// Decoder_39256001085541
// MI455X (gfx1250) — compile-verified, matrix operand order checked by reading, not run on hardware
//
#include <hip/hip_runtime.h>
#include <hip/hip_bf16.h>

// ---------------------------------------------------------------------------
// Tacotron2-style decoder, 200 sequential steps.
// The two [16,K]x[K,N] LSTM gate GEMMs run on v_wmma_f32_16x16x32_f16 with
// f16 weights converted once per call (L2-resident across the loop).
// Key algebraic cut: zero initial LSTM state means the forget gate is dead
// (f * c_prev == 0), so we only compute gates i, g, o -> 25% less weight
// traffic and WMMA work per step. GEMM + activation are fused: 3 waves per
// N-tile (one per live gate), LDS exchange, in-register activation.
// ---------------------------------------------------------------------------

#define BATCH 16
#define TENC  256
#define TDEC  200
#define EDIM  512
#define ATT   128
#define PRE   256
#define HD    1024
#define MEL   80
#define LOC   32
#define KCONV 31

typedef __attribute__((ext_vector_type(16))) _Float16 v16h;
typedef __attribute__((ext_vector_type(8)))  float    v8f;

union HFrag { v16h v; uint4 q[2]; };

__device__ __forceinline__ float sigmoidf_(float x) {
    return 1.f / (1.f + __expf(-x));
}

// ---------------------------------------------------------------------------
// Zero the scan carry (awc, h2) at the start of every call.
// ---------------------------------------------------------------------------
__global__ void init_state_kernel(float* __restrict__ awc, float* __restrict__ h2) {
    int i = blockIdx.x * blockDim.x + threadIdx.x;
    if (i < BATCH * TENC) awc[i] = 0.f;
    if (i < BATCH * HD)   h2[i]  = 0.f;
}

// ---------------------------------------------------------------------------
// fp32 -> f16 conversion of the two big LSTM weight matrices.
// ---------------------------------------------------------------------------
__global__ void cvt_w_kernel(const float* __restrict__ w0, const float* __restrict__ w1,
                             _Float16* __restrict__ w0h, _Float16* __restrict__ w1h) {
    long long idx = (long long)blockIdx.x * blockDim.x + threadIdx.x;
    const long long n0 = 4096LL * (PRE + EDIM);   // w_ih0: [4096, 768]
    const long long n1 = 4096LL * HD;             // w_ih1: [4096, 1024]
    if (idx < n0)            w0h[idx]      = (_Float16)w0[idx];
    else if (idx < n0 + n1)  w1h[idx - n0] = (_Float16)w1[idx - n0];
}

// ---------------------------------------------------------------------------
// Step-invariant: proc_enc[b,t,a] = dot(enc[b,t,:512], enc_proj_w[a,:512]).
// ---------------------------------------------------------------------------
__global__ void proc_enc_kernel(const float* __restrict__ enc,
                                const float* __restrict__ enc_proj_w,
                                float* __restrict__ pe) {
    int bt = blockIdx.x;          // 0..B*TENC-1
    int a  = threadIdx.x;         // 0..127
    const float* e = enc + (long long)bt * EDIM;
    const float* w = enc_proj_w + a * EDIM;
    float acc = 0.f;
    for (int k = 0; k < EDIM; ++k) acc = fmaf(e[k], w[k], acc);
    pe[bt * ATT + a] = acc;
}

// ---------------------------------------------------------------------------
// One attention step: prenet, location-sensitive attention, softmax, context.
// One block per batch element, 256 threads (one per encoder position).
// ---------------------------------------------------------------------------
__global__ void attn_step_kernel(
    const float* __restrict__ enc,          // [16,256,512]
    const float* __restrict__ padded_mels,  // [16,200,80]
    const float* __restrict__ lstm_proj_w,  // [128,1024]
    const float* __restrict__ loc_conv_w,   // [32,31]
    const float* __restrict__ loc_conv_b,   // [32]
    const float* __restrict__ loc_dense_w,  // [128,32]
    const float* __restrict__ loc_dense_b,  // [128]
    const float* __restrict__ e_w,          // [128]
    const float* __restrict__ e_b,          // [1]
    const float* __restrict__ prenet1_w,    // [256,80]
    const float* __restrict__ prenet2_w,    // [256,256]
    const int*   __restrict__ text_len,     // [16]
    const float* __restrict__ proc_enc,     // [16,256,128]
    float*       __restrict__ awc,          // [16,256]  (carry, updated)
    const float* __restrict__ h2,           // [16,1024] (carry from prev step)
    _Float16*    __restrict__ lih,          // [16,768]  f16 LSTM0 input
    float*       __restrict__ ctxf,         // [16,512]
    int t) {
    const int b = blockIdx.x;
    const int tid = threadIdx.x;
    const int lane = tid & 31, wid = tid >> 5;

    __shared__ float s_prev[MEL];
    __shared__ float s_x1[PRE];
    __shared__ float s_q[ATT];
    __shared__ float s_awc[TENC + KCONV - 1];   // padded for the conv halo
    __shared__ float s_aw[TENC];
    __shared__ float s_red[8];
    __shared__ float s_bcast;

    if (tid < MEL)
        s_prev[tid] = (t == 0) ? 0.f
                               : padded_mels[((long long)b * TDEC + (t - 1)) * MEL + tid];
    for (int i = tid; i < TENC + KCONV - 1; i += blockDim.x) {
        int p = i - (KCONV / 2);
        s_awc[i] = (p >= 0 && p < TENC) ? awc[b * TENC + p] : 0.f;
    }
    __syncthreads();

    // prenet layer 1: [80] -> [256], ReLU
    {
        float acc = 0.f;
        const float* w = prenet1_w + tid * MEL;
        for (int k = 0; k < MEL; ++k) acc = fmaf(s_prev[k], w[k], acc);
        s_x1[tid] = fmaxf(acc, 0.f);
    }
    // attention query q = h2 @ lstm_proj_w^T (threads 0..127)
    if (tid < ATT) {
        float acc = 0.f;
        const float* w  = lstm_proj_w + tid * HD;
        const float* hv = h2 + b * HD;
        for (int k = 0; k < HD; ++k) acc = fmaf(hv[k], w[k], acc);
        s_q[tid] = acc;
    }
    __syncthreads();

    // prenet layer 2: [256] -> [256], ReLU; write straight into li[0:256]
    {
        float acc = 0.f;
        const float* w = prenet2_w + tid * PRE;
        for (int k = 0; k < PRE; ++k) acc = fmaf(s_x1[k], w[k], acc);
        float x2 = fmaxf(acc, 0.f);
        lih[b * (PRE + EDIM) + tid] = (_Float16)x2;
    }

    // location conv (K=31, pad 15) at position tid, all 32 channels
    float loc[LOC];
#pragma unroll
    for (int c = 0; c < LOC; ++c) {
        float v = loc_conv_b[c];
        const float* w = loc_conv_w + c * KCONV;
#pragma unroll
        for (int k = 0; k < KCONV; ++k) v = fmaf(s_awc[tid + k], w[k], v);
        loc[c] = v;
    }

    // energy: en[t] = sum_a tanh(q[a] + locdense[a] + proc_enc[a]) * e_w[a]
    float en = e_b[0];
    const float* pe = proc_enc + ((long long)b * TENC + tid) * ATT;
    for (int a = 0; a < ATT; ++a) {
        float la = loc_dense_b[a];
        const float* ldw = loc_dense_w + a * LOC;
#pragma unroll
        for (int c = 0; c < LOC; ++c) la = fmaf(loc[c], ldw[c], la);
        en = fmaf(tanhf(s_q[a] + la + pe[a]), e_w[a], en);
    }
    if (tid >= text_len[b]) en = -1e9f;

    // softmax over 256 positions (wave32 shuffle + LDS, 8 waves)
    float m = en;
    for (int off = 16; off > 0; off >>= 1) m = fmaxf(m, __shfl_xor(m, off, 32));
    if (lane == 0) s_red[wid] = m;
    __syncthreads();
    if (tid == 0) {
        float mm = s_red[0];
        for (int i = 1; i < 8; ++i) mm = fmaxf(mm, s_red[i]);
        s_bcast = mm;
    }
    __syncthreads();
    float ex = __expf(en - s_bcast);
    float s = ex;
    for (int off = 16; off > 0; off >>= 1) s += __shfl_xor(s, off, 32);
    __syncthreads();                 // protect s_bcast before overwrite
    if (lane == 0) s_red[wid] = s;
    __syncthreads();
    if (tid == 0) {
        float ss = 0.f;
        for (int i = 0; i < 8; ++i) ss += s_red[i];
        s_bcast = ss;
    }
    __syncthreads();
    float aw = ex / s_bcast;
    s_aw[tid] = aw;
    awc[b * TENC + tid] += aw;       // carry update (awc + aw)
    __syncthreads();

    // context = aw @ enc[b] : each thread owns 2 of the 512 dims
    float c0 = 0.f, c1 = 0.f;
    const float* eb = enc + (long long)b * TENC * EDIM + tid * 2;
    for (int tt = 0; tt < TENC; ++tt) {
        float w = s_aw[tt];
        c0 = fmaf(w, eb[(long long)tt * EDIM + 0], c0);
        c1 = fmaf(w, eb[(long long)tt * EDIM + 1], c1);
    }
    ctxf[b * EDIM + 2 * tid]     = c0;
    ctxf[b * EDIM + 2 * tid + 1] = c1;
    lih[b * (PRE + EDIM) + PRE + 2 * tid]     = (_Float16)c0;
    lih[b * (PRE + EDIM) + PRE + 2 * tid + 1] = (_Float16)c1;
}

// ---------------------------------------------------------------------------
// Fused LSTM cell (zero state): only gates i, g, o are live.
// Grid: 64 blocks (one per 16-wide tile of the 1024 hidden dims).
// Block: 96 threads = 3 waves; wave w computes gate {i,g,o}[w] via WMMA.
// Waves 1,2 publish their tiles through LDS; wave 0 applies
//   h = sigmoid(o) * tanh( sigmoid(i) * tanh(g) )
// in-register and stores h directly (f32 + optional f16). No gates buffer.
//
// Fragment layout per ISA 7.12.2 (16-bit A 16x32 / B 32x16):
//   lanes 0-15  hold K = k0+{0..7, 16..23},  lanes 16-31 hold K = k0+{8..15, 24..31}.
// ---------------------------------------------------------------------------
template <int KDIM>
__global__ void lstm_fused_kernel(const _Float16* __restrict__ A,    // [16, KDIM]
                                  const _Float16* __restrict__ W,    // [4096, KDIM]
                                  const float* __restrict__ b_ih,
                                  const float* __restrict__ b_hh,
                                  float* __restrict__ hF,            // [16, 1024]
                                  _Float16* __restrict__ hH) {       // [16, 1024] or null
    const int wave = threadIdx.x >> 5;                  // 0:i  1:g  2:o
    const int lane = threadIdx.x & 31;
    const int n0   = blockIdx.x * 16;                   // tile within [0,1024)
    const int r    = lane & 15;          // A row (M) / B column (N) within tile
    const int kSh  = (lane >> 4) * 8;    // half-wave K shift
    // gate row offsets in the [4096, KDIM] weight matrix: i=0, g=2H, o=3H
    const int gateRow = (wave == 0) ? 0 : (wave == 1) ? 2 * HD : 3 * HD;

    __shared__ float s_g[256];    // gate g tile, [m][n]
    __shared__ float s_o[256];    // gate o tile, [m][n]

    const _Float16* arow = A + r * KDIM + kSh;
    const _Float16* wrow = W + (long long)(gateRow + n0 + r) * KDIM + kSh;

    v8f acc = {};
    for (int k0 = 0; k0 < KDIM; k0 += 32) {
        __builtin_prefetch((const void*)(wrow + k0 + 256), 0, 1);
        HFrag a, b;
        a.q[0] = *(const uint4*)(arow + k0);
        a.q[1] = *(const uint4*)(arow + k0 + 16);
        b.q[0] = *(const uint4*)(wrow + k0);
        b.q[1] = *(const uint4*)(wrow + k0 + 16);
        acc = __builtin_amdgcn_wmma_f32_16x16x32_f16(
            /*neg_a=*/false, a.v, /*neg_b=*/false, b.v,
            /*c_mod=*/(short)0, acc, /*reuse_a=*/false, /*reuse_b=*/false);
    }

    // D layout: lanes 0-15 -> N=lane, M=vgpr; lanes 16-31 -> N=lane-16, M=vgpr+8
    const int n = n0 + r;
    const int mbase = (lane >> 4) * 8;
    const float bias = b_ih[gateRow + n] + b_hh[gateRow + n];

    if (wave == 1) {
#pragma unroll
        for (int v = 0; v < 8; ++v) s_g[(mbase + v) * 16 + r] = acc[v] + bias;
    } else if (wave == 2) {
#pragma unroll
        for (int v = 0; v < 8; ++v) s_o[(mbase + v) * 16 + r] = acc[v] + bias;
    }
    __syncthreads();

    if (wave == 0) {
#pragma unroll
        for (int v = 0; v < 8; ++v) {
            const int m  = mbase + v;
            const float gi = acc[v] + bias;
            const float gg = s_g[m * 16 + r];
            const float go = s_o[m * 16 + r];
            const float c  = sigmoidf_(gi) * tanhf(gg);
            const float h  = sigmoidf_(go) * tanhf(c);
            hF[m * HD + n] = h;
            if (hH) hH[m * HD + n] = (_Float16)h;
        }
    }
}

// ---------------------------------------------------------------------------
// Output projection: dec = [h2 (1024), ctx (512)]; mel [80], stop [1].
// One block per batch element; threads 0..79 -> mel dims, thread 80 -> stop.
// ---------------------------------------------------------------------------
__global__ void proj_step_kernel(const float* __restrict__ h2,
                                 const float* __restrict__ ctxf,
                                 const float* __restrict__ proj_w,   // [80,1536]
                                 const float* __restrict__ proj_b,   // [80]
                                 const float* __restrict__ stop_w,   // [1,1536]
                                 const float* __restrict__ stop_b,   // [1]
                                 float* __restrict__ out, int t) {
    int b = blockIdx.x, tid = threadIdx.x;
    if (tid > MEL) return;
    const float* w = (tid < MEL) ? (proj_w + tid * (HD + EDIM)) : stop_w;
    float acc = (tid < MEL) ? proj_b[tid] : stop_b[0];
    const float* hv = h2 + b * HD;
    for (int k = 0; k < HD; ++k) acc = fmaf(hv[k], w[k], acc);
    const float* cv = ctxf + b * EDIM;
    for (int k = 0; k < EDIM; ++k) acc = fmaf(cv[k], w[HD + k], acc);
    if (tid < MEL) out[((long long)b * TDEC + t) * MEL + tid] = acc;
    else           out[(long long)BATCH * TDEC * MEL + b * TDEC + t] = acc;
}

// ---------------------------------------------------------------------------
// Host launcher: prep (init + f16 weights + proc_enc) then 200 decode steps.
// 4 kernels per serial step: attention, LSTM0 (fused), LSTM1 (fused), proj.
// ---------------------------------------------------------------------------
extern "C" void kernel_launch(void* const* d_in, const int* in_sizes, int n_in,
                              void* d_out, int out_size, void* d_ws, size_t ws_size,
                              hipStream_t stream) {
    const float* enc         = (const float*)d_in[0];
    const float* padded_mels = (const float*)d_in[1];
    const float* enc_proj_w  = (const float*)d_in[2];
    const float* lstm_proj_w = (const float*)d_in[3];
    const float* loc_conv_w  = (const float*)d_in[4];
    const float* loc_conv_b  = (const float*)d_in[5];
    const float* loc_dense_w = (const float*)d_in[6];
    const float* loc_dense_b = (const float*)d_in[7];
    const float* e_w         = (const float*)d_in[8];
    const float* e_b         = (const float*)d_in[9];
    const float* prenet1_w   = (const float*)d_in[10];
    const float* prenet2_w   = (const float*)d_in[11];
    const float* w_ih0       = (const float*)d_in[12];
    // d_in[13] = w_hh0 (unused: zero hidden state)
    const float* b_ih0       = (const float*)d_in[14];
    const float* b_hh0       = (const float*)d_in[15];
    const float* w_ih1       = (const float*)d_in[16];
    // d_in[17] = w_hh1 (unused)
    const float* b_ih1       = (const float*)d_in[18];
    const float* b_hh1       = (const float*)d_in[19];
    const float* proj_w      = (const float*)d_in[20];
    const float* proj_b      = (const float*)d_in[21];
    const float* stop_w      = (const float*)d_in[22];
    const float* stop_b      = (const float*)d_in[23];
    const int*   text_len    = (const int*)d_in[24];
    float* out = (float*)d_out;

    // Carve the workspace (256B-aligned slabs).
    char* ws = (char*)d_ws;
    size_t off = 0;
    auto alloc = [&](size_t bytes) -> void* {
        void* p = ws + off;
        off = (off + bytes + 255) & ~(size_t)255;
        return p;
    };
    _Float16* w0h   = (_Float16*)alloc(4096ULL * (PRE + EDIM) * sizeof(_Float16));
    _Float16* w1h   = (_Float16*)alloc(4096ULL * HD * sizeof(_Float16));
    float*    pe    = (float*)alloc((size_t)BATCH * TENC * ATT * sizeof(float));
    float*    awc   = (float*)alloc((size_t)BATCH * TENC * sizeof(float));
    float*    h2    = (float*)alloc((size_t)BATCH * HD * sizeof(float));
    float*    h1f   = (float*)alloc((size_t)BATCH * HD * sizeof(float));
    _Float16* h1h   = (_Float16*)alloc((size_t)BATCH * HD * sizeof(_Float16));
    _Float16* lih   = (_Float16*)alloc((size_t)BATCH * (PRE + EDIM) * sizeof(_Float16));
    float*    ctxf  = (float*)alloc((size_t)BATCH * EDIM * sizeof(float));
    (void)ws_size; (void)in_sizes; (void)n_in; (void)out_size;

    // --- prep ---
    init_state_kernel<<<(BATCH * HD + 255) / 256, 256, 0, stream>>>(awc, h2);
    {
        const long long ncvt = 4096LL * (PRE + EDIM) + 4096LL * HD;
        cvt_w_kernel<<<(int)((ncvt + 255) / 256), 256, 0, stream>>>(w_ih0, w_ih1, w0h, w1h);
    }
    proc_enc_kernel<<<BATCH * TENC, ATT, 0, stream>>>(enc, enc_proj_w, pe);

    // --- 200 sequential decode steps ---
    for (int t = 0; t < TDEC; ++t) {
        attn_step_kernel<<<BATCH, TENC, 0, stream>>>(
            enc, padded_mels, lstm_proj_w, loc_conv_w, loc_conv_b,
            loc_dense_w, loc_dense_b, e_w, e_b, prenet1_w, prenet2_w,
            text_len, pe, awc, h2, lih, ctxf, t);

        // LSTM0: [16,768] x [768,3072 live gates] via WMMA, fused activation
        lstm_fused_kernel<PRE + EDIM><<<HD / 16, 96, 0, stream>>>(
            lih, w0h, b_ih0, b_hh0, h1f, h1h);

        // LSTM1: [16,1024] x [1024,3072 live gates] via WMMA, fused activation
        lstm_fused_kernel<HD><<<HD / 16, 96, 0, stream>>>(
            h1h, w1h, b_ih1, b_hh1, h2, (_Float16*)nullptr);

        proj_step_kernel<<<BATCH, 128, 0, stream>>>(h2, ctxf, proj_w, proj_b,
                                                    stop_w, stop_b, out, t);
    }
}
